// CRFForPreTraining_72550587564485
// MI455X (gfx1250) — matrix operand order repaired and checked by reading, not compiled, not run on hardware
//
#include <hip/hip_runtime.h>
#include <hip/hip_bf16.h>

// Problem constants (from reference): B=256, S=1024, L=128
#define B_ 256
#define S_ 1024
#define L_ 128
#define TSTEPS 1022   // jax.lax.scan length = S-2

typedef __attribute__((ext_vector_type(16))) _Float16 v16h;
typedef __attribute__((ext_vector_type(8)))  _Float16 v8h;
typedef __attribute__((ext_vector_type(8)))  float    v8f;

#define LOG2E_F  1.44269504088896340736f
#define LN2_F    0.69314718055994530942f

// ---- gfx1250 async global->LDS DMA (ASYNCcnt) with graceful fallback -------
#if defined(__has_builtin)
#if __has_builtin(__builtin_amdgcn_global_load_async_to_lds_b128)
#define CDNA5_ASYNC_LDS 1
#endif
#endif
#ifndef CDNA5_ASYNC_LDS
#define CDNA5_ASYNC_LDS 0
#endif

#if CDNA5_ASYNC_LDS
typedef __attribute__((__vector_size__(16))) int v4i;
typedef __attribute__((address_space(1))) v4i gv4i;
typedef __attribute__((address_space(3))) v4i lv4i;
static __device__ __forceinline__ gv4i* to_glob(const void* p) {
  return (gv4i*)(unsigned long long)(uintptr_t)p;
}
static __device__ __forceinline__ lv4i* to_lds(void* p) {
  // flat LDS address keeps the LDS byte offset in its low 32 bits
  return (lv4i*)(unsigned int)(uintptr_t)p;
}
static __device__ __forceinline__ void async_wait0() {
#if __has_builtin(__builtin_amdgcn_s_wait_asynccnt)
  __builtin_amdgcn_s_wait_asynccnt(0);
#else
  asm volatile("s_wait_asynccnt 0x0" ::: "memory");
#endif
}
#endif

// Reduce across the 16-lane half of a wave32 (xor masks < 16 keep halves separate).
__device__ __forceinline__ float half16_max(float v) {
#pragma unroll
  for (int m = 1; m < 16; m <<= 1) v = fmaxf(v, __shfl_xor(v, m, 32));
  return v;
}
__device__ __forceinline__ float half16_sum(float v) {
#pragma unroll
  for (int m = 1; m < 16; m <<= 1) v += __shfl_xor(v, m, 32);
  return v;
}

// ---------------------------------------------------------------------------
// Forward recursion: one workgroup = 16 batch rows, 8 waves = 8 N-slices of 16.
// alpha'[b][j] = m_b + log( sum_i exp(alpha[b][i]-m_b) * exp(T[i][j]) ) + e[b][j]
// Inner sum = 16x128 @ 128x16 f16 WMMA GEMM, f32 accumulate. exp(T) stays in
// VGPRs for the whole sequence; emissions stream once via async LDS DMA.
// ---------------------------------------------------------------------------
__global__ __launch_bounds__(256)
void crf_forward_kernel(const float* __restrict__ emissions,
                        const unsigned char* __restrict__ mask,
                        const float* __restrict__ start_t,
                        const float* __restrict__ end_t,
                        const float* __restrict__ trans,
                        float* __restrict__ logz_out)
{
  __shared__ __align__(16) float    sh_alpha[16 * L_];   // 8 KB recurrent state
  __shared__ __align__(16) _Float16 sh_p[16 * L_];       // 4 KB A matrix stage
  __shared__ __align__(16) float    sh_em[2][16 * L_];   // 16 KB emissions tiles
  __shared__ __align__(16) float sh_mrow[16];
  __shared__ __align__(8) unsigned char sh_mask[16];

  const int tid  = threadIdx.x;
  const int lane = tid & 31;
  const int wave = tid >> 5;                 // 0..7 -> owns columns [16w,16w+16)
  const int b0   = blockIdx.x * 16;          // batch tile base
  const int r    = tid >> 4;                 // 0..15: row for load/exp phases
  const int c8   = (tid & 15) * 8;           // 8-col strip per thread
  const int n    = wave * 16 + (lane & 15);  // owned N column
  const int mb   = (lane >> 4) * 8;          // C-fragment row base

  // -------- Pack resident B fragments: exp(trans), 128x16 slice per wave ----
  v16h bfrag[4];
  {
    const int kb = (lane >> 4) * 16;
#pragma unroll
    for (int c = 0; c < 4; ++c) {
#pragma unroll
      for (int v = 0; v < 8; ++v) {
        const int k = 32 * c + kb + 2 * v;
        bfrag[c][2 * v] =
            (_Float16)__builtin_amdgcn_exp2f(trans[k * L_ + n] * LOG2E_F);
        bfrag[c][2 * v + 1] =
            (_Float16)__builtin_amdgcn_exp2f(trans[(k + 1) * L_ + n] * LOG2E_F);
      }
    }
  }

  // -------- alpha0 = start_t + emissions[:,1,:] ------------------------------
  {
    const size_t base = ((size_t)(b0 + r) * S_ + 1) * L_ + c8;
#pragma unroll
    for (int j = 0; j < 8; ++j)
      sh_alpha[r * L_ + c8 + j] = start_t[c8 + j] + emissions[base + j];
  }
  // -------- preload emissions tile for step 0 (t = 2) ------------------------
  {
    const size_t base = ((size_t)(b0 + r) * S_ + 2) * L_ + c8;
#if CDNA5_ASYNC_LDS
    __builtin_amdgcn_global_load_async_to_lds_b128(
        to_glob(emissions + base),     to_lds(&sh_em[0][r * L_ + c8]),     0, 0);
    __builtin_amdgcn_global_load_async_to_lds_b128(
        to_glob(emissions + base + 4), to_lds(&sh_em[0][r * L_ + c8 + 4]), 0, 0);
    async_wait0();
#else
    float4 a = *(const float4*)(emissions + base);
    float4 b = *(const float4*)(emissions + base + 4);
    *(float4*)&sh_em[0][r * L_ + c8]     = a;
    *(float4*)&sh_em[0][r * L_ + c8 + 4] = b;
#endif
  }
  __syncthreads();

  // Wave-resident alpha fragment at C positions (m = mb+v, col n).
  float areg[8];
#pragma unroll
  for (int v = 0; v < 8; ++v) areg[v] = sh_alpha[(mb + v) * L_ + n];

  // Running source pointer for the tile of step k+1 (t = k+3).
  const float* gnext = emissions + ((size_t)(b0 + r) * S_ + 3) * L_ + c8;
  const unsigned char* mrun = mask + (size_t)(b0 + (tid & 15)) * S_ + 2;

  int cur = 0;
  for (int k = 0; k < TSTEPS; ++k) {
    // ---- phase 1: row max, exp (raw v_exp_f32), stage A matrix in f16 ----
    {
      float4 a0 = *(const float4*)&sh_alpha[r * L_ + c8];
      float4 a1 = *(const float4*)&sh_alpha[r * L_ + c8 + 4];
      float al[8] = {a0.x, a0.y, a0.z, a0.w, a1.x, a1.y, a1.z, a1.w};
      float vmax = al[0];
#pragma unroll
      for (int j = 1; j < 8; ++j) vmax = fmaxf(vmax, al[j]);
      const float m_r = half16_max(vmax);
      v8h ph;
#pragma unroll
      for (int j = 0; j < 8; ++j)
        ph[j] = (_Float16)__builtin_amdgcn_exp2f((al[j] - m_r) * LOG2E_F);
      *(v8h*)&sh_p[r * L_ + c8] = ph;
      if ((tid & 15) == 0) sh_mrow[r] = m_r;
    }
    if (tid < 16)
      sh_mask[tid] = mrun[k];

    // ---- phase 2: stage next emissions tile (t = k+3) ----
#if CDNA5_ASYNC_LDS
    if (k + 1 < TSTEPS) {
      float* dst = &sh_em[cur ^ 1][r * L_ + c8];
      __builtin_amdgcn_global_load_async_to_lds_b128(
          to_glob(gnext),     to_lds(dst),     0, 0);
      __builtin_amdgcn_global_load_async_to_lds_b128(
          to_glob(gnext + 4), to_lds(dst + 4), 0, 0);
    }
#else
    float4 ra = {}, rb = {};
    if (k + 1 < TSTEPS) {
      ra = *(const float4*)(gnext);
      rb = *(const float4*)(gnext + 4);
    }
#endif
    if (k + 6 < TSTEPS)   // warm L2 ahead (global_prefetch_b8, imm offset)
      __builtin_prefetch(gnext + 5 * L_, 0, 1);
    __syncthreads();

    // ---- phase 3: GEMM slice via v_wmma_f32_16x16x32_f16, K = 128 ----
    v8f acc = {};
    {
      const int m  = lane & 15;
      const int kh = (lane >> 4) * 8;
#pragma unroll
      for (int c = 0; c < 4; ++c) {
        v16h af;  // A 16x32 f16 layout
#pragma unroll
        for (int v = 0; v < 8; ++v) {
          const int kk = 32 * c + kh * 2 + 2 * v;
          af[2 * v]     = sh_p[m * L_ + kk];
          af[2 * v + 1] = sh_p[m * L_ + kk + 1];
        }
        acc = __builtin_amdgcn_wmma_f32_16x16x32_f16(
                  false, af, false, bfrag[c], (short)0, acc, false, false);
      }
    }

    // ---- phase 4a: unconditionally compute all 8 candidate alphas ----
    float nxt[8];
    {
      float4 mr0 = *(const float4*)&sh_mrow[mb];      // ds_load_b128
      float4 mr1 = *(const float4*)&sh_mrow[mb + 4];  // ds_load_b128
      const float mr[8] = {mr0.x, mr0.y, mr0.z, mr0.w,
                           mr1.x, mr1.y, mr1.z, mr1.w};
#pragma unroll
      for (int v = 0; v < 8; ++v) {
        // acc >= exp(-0.1): raw v_log_f32 (log2) is safe, no denorm fixup
        nxt[v] = mr[v] + __builtin_amdgcn_logf(acc[v]) * LN2_F +
                 sh_em[cur][(mb + v) * L_ + n];
      }
    }
    // ---- phase 4b: branchless arithmetic blend, write back ----
    {
      const unsigned int* mw = (const unsigned int*)sh_mask;
      const unsigned int w0 = mw[mb >> 2];
      const unsigned int w1 = mw[(mb >> 2) + 1];
#pragma unroll
      for (int v = 0; v < 8; ++v) {
        const unsigned int mbyte = ((v < 4 ? w0 : w1) >> ((v & 3) * 8)) & 0xffu;
        const float mf = mbyte ? 1.0f : 0.0f;
        areg[v] = fmaf(mf, nxt[v] - areg[v], areg[v]);
        sh_alpha[(mb + v) * L_ + n] = areg[v];
      }
    }

    // ---- phase 5: make next tile visible, close the step ----
#if CDNA5_ASYNC_LDS
    async_wait0();
#else
    if (k + 1 < TSTEPS) {
      *(float4*)&sh_em[cur ^ 1][r * L_ + c8]     = ra;
      *(float4*)&sh_em[cur ^ 1][r * L_ + c8 + 4] = rb;
    }
#endif
    __syncthreads();
    cur ^= 1;
    gnext += L_;
  }

  // -------- log_z = logsumexp(alpha + end_t) per batch row -------------------
  {
    float ev[8];
    float vmax = -3.4e38f;
#pragma unroll
    for (int j = 0; j < 8; ++j) {
      ev[j] = sh_alpha[r * L_ + c8 + j] + end_t[c8 + j];
      vmax = fmaxf(vmax, ev[j]);
    }
    const float m_r = half16_max(vmax);
    float s = 0.f;
#pragma unroll
    for (int j = 0; j < 8; ++j)
      s += __builtin_amdgcn_exp2f((ev[j] - m_r) * LOG2E_F);
    s = half16_sum(s);
    if ((tid & 15) == 0)
      logz_out[b0 + r] = m_r + __builtin_amdgcn_logf(s) * LN2_F;
  }
}

// ---------------------------------------------------------------------------
// Numerator: one wave per batch; lanes stride the time axis (pure gathers).
// ---------------------------------------------------------------------------
__global__ __launch_bounds__(256)
void crf_num_kernel(const float* __restrict__ emissions,
                    const int* __restrict__ labels,
                    const unsigned char* __restrict__ mask,
                    const float* __restrict__ start_t,
                    const float* __restrict__ end_t,
                    const float* __restrict__ trans,
                    float* __restrict__ num_out)
{
  const int lane = threadIdx.x & 31;
  const int wave = threadIdx.x >> 5;
  const int b    = blockIdx.x * 8 + wave;
  if (b >= B_) return;

  float acc = 0.f;
  int   cnt = 0;
  for (int k = lane; k < TSTEPS; k += 32) {
    const int tagk  = labels[b * S_ + k + 1];
    const int tagk1 = labels[b * S_ + k + 2];
    const float mk  = (float)mask[(size_t)b * S_ + k + 2];
    acc += (trans[tagk * L_ + tagk1] +
            emissions[((size_t)b * S_ + k + 2) * L_ + tagk1]) * mk;
  }
  for (int i = lane; i < S_ - 1; i += 32)
    cnt += (int)mask[(size_t)b * S_ + 1 + i];
#pragma unroll
  for (int m = 1; m < 32; m <<= 1) {
    acc += __shfl_xor(acc, m, 32);
    cnt += __shfl_xor(cnt, m, 32);
  }
  if (lane == 0) {
    const int tag0 = labels[b * S_ + 1];
    const int last = labels[b * S_ + 1 + (cnt - 1)];
    num_out[b] = start_t[tag0] +
                 emissions[((size_t)b * S_ + 1) * L_ + tag0] +
                 acc + end_t[last];
  }
}

// ---------------------------------------------------------------------------
// Final deterministic reduction: out = -sum_b(num[b] - logz[b])
// ---------------------------------------------------------------------------
__global__ __launch_bounds__(256)
void crf_reduce_kernel(const float* __restrict__ num,
                       const float* __restrict__ logz,
                       float* __restrict__ out)
{
  __shared__ float sh[256];
  const int t = threadIdx.x;
  sh[t] = num[t] - logz[t];
  __syncthreads();
  for (int s = 128; s > 0; s >>= 1) {
    if (t < s) sh[t] += sh[t + s];
    __syncthreads();
  }
  if (t == 0) out[0] = -sh[0];
}

extern "C" void kernel_launch(void* const* d_in, const int* in_sizes, int n_in,
                              void* d_out, int out_size, void* d_ws, size_t ws_size,
                              hipStream_t stream) {
  const float*         emissions = (const float*)d_in[0];
  const int*           labels    = (const int*)d_in[1];
  const unsigned char* mask      = (const unsigned char*)d_in[2];
  const float*         start_t   = (const float*)d_in[3];
  const float*         end_t     = (const float*)d_in[4];
  const float*         trans     = (const float*)d_in[5];
  float* out  = (float*)d_out;
  float* logz = (float*)d_ws;          // [256]
  float* num  = logz + B_;             // [256]

  crf_forward_kernel<<<B_ / 16, 256, 0, stream>>>(emissions, mask, start_t,
                                                  end_t, trans, logz);
  crf_num_kernel<<<B_ / 8, 256, 0, stream>>>(emissions, labels, mask, start_t,
                                             end_t, trans, num);
  crf_reduce_kernel<<<1, 256, 0, stream>>>(num, logz, out);
}
